// BigramAttn_74242804679157
// MI455X (gfx1250) — matrix operand order repaired and checked by reading, not compiled, not run on hardware
//
#include <hip/hip_runtime.h>

// ---------------------------------------------------------------------------
// BigramAttn for MI455X (gfx1250, wave32, WMMA)
//   e[b,s] = (enc[s-1,b] @ W) . (enc[s,b] * h[b])   (s>0;  s==0: enc[0,b].h[b])
//          + (h[b] @ affect) . emb[s,b]
//   out[b,1,s] = softmax_s(e[b,s])
// Dominant GEMM: v_wmma_f32_16x16x32_bf16 with persistent per-tile f32
// accumulators (D->C chaining, no hazard), g-fold epilogue run once.
// ---------------------------------------------------------------------------

typedef __attribute__((ext_vector_type(16))) __bf16          v16bf;
typedef __attribute__((ext_vector_type(8)))  float           v8f;
typedef __attribute__((ext_vector_type(8)))  unsigned short  v8u16;
typedef __attribute__((ext_vector_type(16))) unsigned short  v16u16;

__device__ __forceinline__ unsigned short f2bf(float f) {
  unsigned u = __float_as_uint(f);
  u += 0x7FFFu + ((u >> 16) & 1u);          // round-to-nearest-even
  return (unsigned short)(u >> 16);
}
__device__ __forceinline__ float bf2f(unsigned short h) {
  return __uint_as_float(((unsigned)h) << 16);
}

static constexpr int Sn = 4096;
static constexpr int Bn = 32;
static constexpr int Hn = 512;

// ---------------------------------------------------------------------------
// prep: W[H,H] f32  ->  Wt[n,k] bf16 (transposed: B-frag lane n, half h
// holds K = h*8 .. h*8+7 and 16+h*8 .. 16+h*8+7)
// ---------------------------------------------------------------------------
__global__ __launch_bounds__(256)
void prep_wt(const float* __restrict__ w, unsigned short* __restrict__ wt) {
  int o = blockIdx.x * 256 + threadIdx.x;   // 0 .. 512*512-1
  int n = o >> 9, k = o & 511;
  wt[o] = f2bf(w[k * Hn + n]);
}

// ---------------------------------------------------------------------------
// prep: per-batch  e[b,0] = enc[0,b].h[b]   and   haff[b,:] = h[b] @ affect
// ---------------------------------------------------------------------------
__global__ __launch_bounds__(256)
void prep_row0(const float* __restrict__ enc, const float* __restrict__ hid,
               const float* __restrict__ aff, float* __restrict__ ews,
               float* __restrict__ haff) {
  __shared__ float red[256];
  const int b = blockIdx.x, tid = threadIdx.x;
  float d0 = 0.f, a0 = 0.f, a1 = 0.f, a2 = 0.f;
  for (int k = tid; k < Hn; k += 256) {
    float hv = hid[b * Hn + k];
    d0 += enc[b * Hn + k] * hv;             // enc[0,b,k]
    a0 += hv * aff[k * 3 + 0];
    a1 += hv * aff[k * 3 + 1];
    a2 += hv * aff[k * 3 + 2];
  }
  float vals[4] = {d0, a0, a1, a2};
  for (int j = 0; j < 4; ++j) {
    red[tid] = vals[j];
    __syncthreads();
    for (int off = 128; off; off >>= 1) {
      if (tid < off) red[tid] += red[tid + off];
      __syncthreads();
    }
    if (tid == 0) {
      if (j == 0) ews[(long)b * Sn] = red[0];
      else        haff[b * 3 + (j - 1)] = red[0];
    }
    __syncthreads();
  }
}

// ---------------------------------------------------------------------------
// main: 64 rows (r = s*B+b) per block, 8 waves each own a 64-wide N slice.
// Hot loop = pure loads + 16 WMMA per 32-wide k-chunk into persistent
// accumulators; the linear g-fold + cross-lane/cross-wave reduction runs once.
// ---------------------------------------------------------------------------
__global__ __launch_bounds__(256)
void bigram_main(const float* __restrict__ enc, const float* __restrict__ hid,
                 const unsigned short* __restrict__ wt, float* __restrict__ ews) {
  constexpr int AP = 520;                   // padded u16 row stride (bank-safe)
  __shared__ unsigned short As[64 * AP];    // A panel: enc[r-B] in bf16
  __shared__ unsigned short Gs[64 * AP];    // g panel: enc[r]*h[b] in bf16
  __shared__ float edot[64];

  const int tid  = threadIdx.x;
  const int wave = tid >> 5;
  const int lane = tid & 31;
  const int l16  = lane & 15;
  const int half = lane >> 4;
  const long rbase = (long)blockIdx.x * 64;

  if (tid < 64) edot[tid] = 0.0f;

  // ---- stage panels, float4-vectorized (enc contiguous in r = s*B+b)
  for (int i = 0; i < 32; ++i) {
    int idx = (i * 256 + tid) * 4;          // 0..32764, 4 consecutive k
    int m = idx >> 9;                       // row-in-tile 0..63
    int k = idx & 511;
    long r = rbase + m;
    float4 av = make_float4(0.f, 0.f, 0.f, 0.f);
    if (r >= Bn) av = *(const float4*)&enc[(r - Bn) * Hn + k];
    const float4 gv = *(const float4*)&enc[r * Hn + k];
    const float4 hv = *(const float4*)&hid[(int)(r & (Bn - 1)) * Hn + k];
    ushort4 a4, g4;
    a4.x = f2bf(av.x); a4.y = f2bf(av.y); a4.z = f2bf(av.z); a4.w = f2bf(av.w);
    g4.x = f2bf(gv.x * hv.x); g4.y = f2bf(gv.y * hv.y);
    g4.z = f2bf(gv.z * hv.z); g4.w = f2bf(gv.w * hv.w);
    *(ushort4*)&As[m * AP + k] = a4;
    *(ushort4*)&Gs[m * AP + k] = g4;
  }
  __syncthreads();

  const int nw = wave * 64;

  // persistent accumulators: 4 N-tiles x 4 M-tiles, 8 f32 each (128 VGPRs)
  v8f acc[4][4];
  #pragma unroll
  for (int nt = 0; nt < 4; ++nt)
    #pragma unroll
    for (int mt = 0; mt < 4; ++mt) acc[nt][mt] = (v8f){};

  for (int k0 = 0; k0 < 512; k0 += 32) {
    // A fragments: lane l16 = row; half picks K chunks {h*8.., 16+h*8..}
    v16bf afr[4];
    #pragma unroll
    for (int mt = 0; mt < 4; ++mt) {
      const unsigned short* ap = &As[(mt * 16 + l16) * AP + k0 + half * 8];
      v8u16 lo = *(const v8u16*)ap;
      v8u16 hi = *(const v8u16*)(ap + 16);
      v16u16 t;
      #pragma unroll
      for (int j = 0; j < 8; ++j) { t[j] = lo[j]; t[j + 8] = hi[j]; }
      afr[mt] = __builtin_bit_cast(v16bf, t);
    }
    #pragma unroll
    for (int nt = 0; nt < 4; ++nt) {
      const int n = nw + nt * 16 + l16;     // B: lane = col, K chunks {h*8.., 16+h*8..}
      const unsigned short* bp = &wt[(long)n * Hn + k0 + half * 8];
      v8u16 blo = *(const v8u16*)bp;
      v8u16 bhi = *(const v8u16*)(bp + 16);
      v16u16 bu;
      #pragma unroll
      for (int j = 0; j < 8; ++j) { bu[j] = blo[j]; bu[j + 8] = bhi[j]; }
      v16bf bfr = __builtin_bit_cast(v16bf, bu);
      #pragma unroll
      for (int mt = 0; mt < 4; ++mt)
        acc[nt][mt] = __builtin_amdgcn_wmma_f32_16x16x32_bf16(
            false, afr[mt], false, bfr, (short)0, acc[nt][mt], false, false);
    }
  }

  // ---- epilogue (once): fold y-tiles into per-row scalars via g
  float val[4][8];
  #pragma unroll
  for (int mt = 0; mt < 4; ++mt)
    #pragma unroll
    for (int v = 0; v < 8; ++v) val[mt][v] = 0.0f;

  #pragma unroll
  for (int nt = 0; nt < 4; ++nt)
    #pragma unroll
    for (int mt = 0; mt < 4; ++mt)
      #pragma unroll
      for (int v = 0; v < 8; ++v) {         // C layout: row = v + 8*half
        int m = mt * 16 + v + 8 * half;
        val[mt][v] += acc[nt][mt][v] * bf2f(Gs[m * AP + nw + nt * 16 + l16]);
      }

  // reduce over the 16 lanes of each half (cols), then across waves via LDS
  #pragma unroll
  for (int mt = 0; mt < 4; ++mt)
    #pragma unroll
    for (int v = 0; v < 8; ++v) {
      float x = val[mt][v];
      x += __shfl_xor(x, 8, 16);
      x += __shfl_xor(x, 4, 16);
      x += __shfl_xor(x, 2, 16);
      x += __shfl_xor(x, 1, 16);
      if (l16 == 0) atomicAdd(&edot[mt * 16 + v + 8 * half], x);
    }
  __syncthreads();

  if (tid < 64) {
    long r = rbase + tid;
    if (r >= Bn) {                          // s==0 handled by prep_row0
      int s = (int)(r >> 5);
      int b = (int)(r & 31);
      ews[(long)b * Sn + s] = edot[tid];
    }
  }
}

// ---------------------------------------------------------------------------
// softmax over s (4096) per batch, fusing the affect term
// ---------------------------------------------------------------------------
__global__ __launch_bounds__(256)
void softmax_out(const float* __restrict__ ews, const float* __restrict__ haff,
                 const float* __restrict__ emb, float* __restrict__ out) {
  __shared__ float red[256];
  const int b = blockIdx.x, tid = threadIdx.x;
  const float h0 = haff[b * 3 + 0], h1 = haff[b * 3 + 1], h2 = haff[b * 3 + 2];
  float ev[16];
  float mx = -3.4e38f;
  #pragma unroll
  for (int i = 0; i < 16; ++i) {
    int s = i * 256 + tid;
    const float* em = &emb[((long)s * Bn + b) * 3];
    float e = ews[(long)b * Sn + s] + h0 * em[0] + h1 * em[1] + h2 * em[2];
    ev[i] = e;
    mx = fmaxf(mx, e);
  }
  red[tid] = mx; __syncthreads();
  for (int off = 128; off; off >>= 1) {
    if (tid < off) red[tid] = fmaxf(red[tid], red[tid + off]);
    __syncthreads();
  }
  mx = red[0]; __syncthreads();
  float sum = 0.f;
  #pragma unroll
  for (int i = 0; i < 16; ++i) { ev[i] = __expf(ev[i] - mx); sum += ev[i]; }
  red[tid] = sum; __syncthreads();
  for (int off = 128; off; off >>= 1) {
    if (tid < off) red[tid] += red[tid + off];
    __syncthreads();
  }
  const float inv = 1.0f / red[0];
  #pragma unroll
  for (int i = 0; i < 16; ++i)
    out[(long)b * Sn + i * 256 + tid] = ev[i] * inv;   // out[B,1,S]
}

// ---------------------------------------------------------------------------
extern "C" void kernel_launch(void* const* d_in, const int* in_sizes, int n_in,
                              void* d_out, int out_size, void* d_ws, size_t ws_size,
                              hipStream_t stream) {
  const float* hid = (const float*)d_in[0];   // [1,B,H]
  const float* enc = (const float*)d_in[1];   // [S,B,H]
  const float* emb = (const float*)d_in[2];   // [S,B,3]
  const float* W   = (const float*)d_in[3];   // [H,H]
  const float* aff = (const float*)d_in[4];   // [H,3]
  float* out = (float*)d_out;

  // workspace layout: Wt bf16 (512 KB) | e scores f32 (512 KB) | haff (384 B)
  unsigned short* wt  = (unsigned short*)d_ws;
  float*          ews = (float*)((char*)d_ws + 512 * 1024);
  float*          ha  = (float*)((char*)d_ws + 1024 * 1024);

  prep_wt    <<<1024, 256, 0, stream>>>(W, wt);
  prep_row0  <<<  32, 256, 0, stream>>>(enc, hid, aff, ews, ha);
  bigram_main<<<2048, 256, 0, stream>>>(enc, hid, wt, ews);
  softmax_out<<<  32, 256, 0, stream>>>(ews, ha, emb, out);
}
